// Attention2d_72782515798250
// MI455X (gfx1250) — matrix operand order repaired and checked by reading, not compiled, not run on hardware
//
#include <hip/hip_runtime.h>

// ---------------------------------------------------------------------------
// Attention2d for MI455X (gfx1250, wave32, WMMA f32_16x16x32_f16).
// B=2, C=128, H=W=64 (T=4096), heads=4, ch=32. Flash attention.
// Attention wave owns 32 query rows: every K/V B-fragment feeds 2 WMMAs.
// ---------------------------------------------------------------------------

typedef __attribute__((ext_vector_type(16))) _Float16 v16h;
typedef __attribute__((ext_vector_type(8)))  float    v8f;

#define NB   2
#define NC   128
#define TT   4096
#define NH   4
#define CH   32
#define BH   (NB*NH)          // 8
#define QK_SCALE 0.42044820762685725f   // 1/sqrt(sqrt(32))

union Frag16 { v16h v; _Float16 e[16]; unsigned u[8]; uint4 q[2]; };

// --- f32 -> f16 convert (weights) ------------------------------------------
__global__ void cvt_f32_f16_kernel(const float* __restrict__ s,
                                   _Float16* __restrict__ d, int n) {
  int i = blockIdx.x * 256 + threadIdx.x;
  if (i < n) d[i] = (_Float16)s[i];
}

// --- x[b][c][t] f32  ->  xT[b][t][c] f16  (LDS-tiled transpose) ------------
__global__ __launch_bounds__(256)
void transpose_x_kernel(const float* __restrict__ x, _Float16* __restrict__ xT) {
  __shared__ _Float16 tile[32][33];
  const int tx = threadIdx.x, ty = threadIdx.y;
  int blk = blockIdx.x;
  const int b = blk / ((TT / 32) * (NC / 32));
  blk %= (TT / 32) * (NC / 32);
  const int c0 = (blk / (TT / 32)) * 32;
  const int t0 = (blk % (TT / 32)) * 32;
  const float* src = x + ((size_t)b * NC + c0) * TT + t0;
  #pragma unroll
  for (int i = 0; i < 4; ++i)
    tile[ty + i * 8][tx] = (_Float16)src[(size_t)(ty + i * 8) * TT + tx];
  __syncthreads();
  _Float16* dst = xT + ((size_t)b * TT + t0) * NC + c0;
  #pragma unroll
  for (int i = 0; i < 4; ++i)
    dst[(size_t)(ty + i * 8) * NC + tx] = tile[tx][ty + i * 8];
}

// --- QKV projection: one wave = 16(o) x 16(t) tile, K=128 in 4x32 ----------
__global__ __launch_bounds__(32)
void qkv_kernel(const _Float16* __restrict__ xT,   // [b][t][c] f16
                const _Float16* __restrict__ wq,   // [384][128] f16
                const float*    __restrict__ bq,   // [384]
                _Float16* __restrict__ Qb,         // [bh][t][ch] (scaled)
                _Float16* __restrict__ Kb,         // [bh][t][ch] (scaled)
                _Float16* __restrict__ Vb) {       // [bh][ch][t]  (transposed)
  const int lane = threadIdx.x;
  const int n = lane & 15, g = lane >> 4;
  int blk = blockIdx.x;
  const int b  = blk / (24 * 256);
  blk %= (24 * 256);
  const int o0 = (blk >> 8) * 16;
  const int t0 = (blk & 255) * 16;

  v8f acc;
  #pragma unroll
  for (int r = 0; r < 8; ++r) acc[r] = bq[o0 + g * 8 + r];

  const _Float16* xrow = xT + ((size_t)b * TT + t0 + n) * NC;
  #pragma unroll
  for (int kk = 0; kk < 4; ++kk) {
    Frag16 a;
    const unsigned* ap = (const unsigned*)(wq + (size_t)(o0 + n) * NC + kk * 32);
    #pragma unroll
    for (int i = 0; i < 8; ++i) {
      int h2 = i >> 2, q = i & 3;
      a.u[i] = ap[g * 8 + h2 * 4 + q];
    }
    Frag16 bf;
    const uint4* bp = (const uint4*)(xrow + kk * 32 + g * 16);
    bf.q[0] = bp[0];
    bf.q[1] = bp[1];
    acc = __builtin_amdgcn_wmma_f32_16x16x32_f16(false, a.v, false, bf.v,
                                                 (short)0, acc, false, false);
  }

  const int j0   = o0 % 96;
  const int seg  = j0 / 32;            // 0=q 1=k 2=v
  const int head = o0 / 96;
  const int cc0  = j0 - seg * 32;
  const int bh   = b * NH + head;
  if (seg == 2) {  // V: [bh][c][t]
    _Float16* p = Vb + ((size_t)bh * CH + cc0 + g * 8) * TT + t0 + n;
    #pragma unroll
    for (int r = 0; r < 8; ++r) p[(size_t)r * TT] = (_Float16)acc[r];
  } else {         // Q/K: [bh][t][c]
    _Float16* dst = (seg == 0) ? Qb : Kb;
    _Float16* p = dst + ((size_t)bh * TT + t0 + n) * CH + cc0 + g * 8;
    #pragma unroll
    for (int r = 0; r < 8; ++r) p[r] = (_Float16)(acc[r] * QK_SCALE);
  }
}

// --- Flash attention: 4 waves/block; each wave owns 32 query rows ----------
__global__ __launch_bounds__(128)
void attn_kernel(const _Float16* __restrict__ Qb,   // [bh][t][ch]
                 const _Float16* __restrict__ Kb,   // [bh][t][ch]
                 const _Float16* __restrict__ Vb,   // [bh][ch][t]
                 _Float16* __restrict__ Ab) {       // [b][t][c] f16
  __shared__ __attribute__((aligned(16))) _Float16 pLds[4][32 * 36];
  const int wid  = threadIdx.x >> 5;
  const int lane = threadIdx.x & 31;
  const int n = lane & 15, g = (lane >> 4) & 1;
  const int tileIdx = blockIdx.x * 4 + wid;       // 128 row-tiles per head
  const int bh = tileIdx >> 7;
  const int t0 = (tileIdx & 127) * 32;
  _Float16* pl = pLds[wid];

  // Two Q fragments (A layout): rows t0..t0+15 and t0+16..t0+31
  Frag16 aq[2];
  #pragma unroll
  for (int h = 0; h < 2; ++h) {
    const unsigned* qp =
        (const unsigned*)(Qb + ((size_t)bh * TT + t0 + h * 16 + n) * CH);
    #pragma unroll
    for (int i = 0; i < 8; ++i) {
      int h2 = i >> 2, q = i & 3;
      aq[h].u[i] = qp[g * 8 + h2 * 4 + q];
    }
  }

  v8f ov[2][2];                       // [rowgrp][colgrp(c 0-15 / 16-31)]
  float mrow[2][8], lrow[2][8];
  #pragma unroll
  for (int h = 0; h < 2; ++h) {
    ov[h][0] = (v8f){0,0,0,0,0,0,0,0};
    ov[h][1] = (v8f){0,0,0,0,0,0,0,0};
    #pragma unroll
    for (int r = 0; r < 8; ++r) { mrow[h][r] = -1e30f; lrow[h][r] = 0.f; }
  }

  const _Float16* kbase = Kb + (size_t)bh * TT * CH;
  const _Float16* vbase = Vb + (size_t)bh * CH * TT;

  for (int s0 = 0; s0 < TT; s0 += 32) {
    if (s0 + 32 < TT) {
      __builtin_prefetch(kbase + (size_t)(s0 + 32 + n) * CH, 0, 1);
      __builtin_prefetch(vbase + (size_t)n * TT + s0 + 32, 0, 1);
    }

    // K^T fragments (B layout), shared by both row groups
    Frag16 bk0, bk1;
    {
      const uint4* kp0 = (const uint4*)(kbase + (size_t)(s0 + n) * CH + g * 16);
      const uint4* kp1 = (const uint4*)(kbase + (size_t)(s0 + 16 + n) * CH + g * 16);
      bk0.q[0] = kp0[0]; bk0.q[1] = kp0[1];
      bk1.q[0] = kp1[0]; bk1.q[1] = kp1[1];
    }

    v8f sv[2][2];
    #pragma unroll
    for (int h = 0; h < 2; ++h) {
      sv[h][0] = (v8f){0,0,0,0,0,0,0,0};
      sv[h][1] = (v8f){0,0,0,0,0,0,0,0};
      sv[h][0] = __builtin_amdgcn_wmma_f32_16x16x32_f16(
          false, aq[h].v, false, bk0.v, (short)0, sv[h][0], false, false);
      sv[h][1] = __builtin_amdgcn_wmma_f32_16x16x32_f16(
          false, aq[h].v, false, bk1.v, (short)0, sv[h][1], false, false);
    }

    // Online softmax per row group; xor-shuffles 1,2,4,8 reduce one row.
    #pragma unroll
    for (int h = 0; h < 2; ++h) {
      #pragma unroll
      for (int r = 0; r < 8; ++r) {
        float a0 = sv[h][0][r], a1 = sv[h][1][r];
        float mb = fmaxf(a0, a1);
        mb = fmaxf(mb, __shfl_xor(mb, 1, 32));
        mb = fmaxf(mb, __shfl_xor(mb, 2, 32));
        mb = fmaxf(mb, __shfl_xor(mb, 4, 32));
        mb = fmaxf(mb, __shfl_xor(mb, 8, 32));
        float mnew  = fmaxf(mrow[h][r], mb);
        float alpha = __expf(mrow[h][r] - mnew);
        float p0 = __expf(a0 - mnew);
        float p1 = __expf(a1 - mnew);
        float rs = p0 + p1;
        rs += __shfl_xor(rs, 1, 32);
        rs += __shfl_xor(rs, 2, 32);
        rs += __shfl_xor(rs, 4, 32);
        rs += __shfl_xor(rs, 8, 32);
        lrow[h][r] = lrow[h][r] * alpha + rs;
        mrow[h][r] = mnew;
        ov[h][0][r] *= alpha;
        ov[h][1][r] *= alpha;
        pl[(h * 16 + g * 8 + r) * 36 + n]      = (_Float16)p0;
        pl[(h * 16 + g * 8 + r) * 36 + 16 + n] = (_Float16)p1;
      }
    }
    asm volatile("s_wait_dscnt 0" ::: "memory");  // P visible before reload

    // V^T fragments (B layout), shared by both row groups
    Frag16 bv0, bv1;
    {
      const uint4* vp0 = (const uint4*)(vbase + (size_t)n * TT + s0 + g * 16);
      const uint4* vp1 = (const uint4*)(vbase + (size_t)(16 + n) * TT + s0 + g * 16);
      bv0.q[0] = vp0[0]; bv0.q[1] = vp0[1];
      bv1.q[0] = vp1[0]; bv1.q[1] = vp1[1];
    }

    #pragma unroll
    for (int h = 0; h < 2; ++h) {
      // Reload P rows for this group in A layout (same-wave LDS is in-order)
      Frag16 apv;
      const unsigned* pp = (const unsigned*)(pl + (h * 16 + n) * 36);
      #pragma unroll
      for (int i = 0; i < 8; ++i) {
        int h2 = i >> 2, q = i & 3;
        apv.u[i] = pp[g * 8 + h2 * 4 + q];
      }
      ov[h][0] = __builtin_amdgcn_wmma_f32_16x16x32_f16(
          false, apv.v, false, bv0.v, (short)0, ov[h][0], false, false);
      ov[h][1] = __builtin_amdgcn_wmma_f32_16x16x32_f16(
          false, apv.v, false, bv1.v, (short)0, ov[h][1], false, false);
    }
  }

  // Normalize, store attn output as f16 in [b][t][c]
  const int b = bh >> 2, head = bh & 3;
  #pragma unroll
  for (int h = 0; h < 2; ++h) {
    _Float16* outp =
        Ab + ((size_t)b * TT + t0 + h * 16 + g * 8) * NC + head * CH + n;
    #pragma unroll
    for (int r = 0; r < 8; ++r) {
      float inv = 1.0f / lrow[h][r];
      outp[(size_t)r * NC]      = (_Float16)(ov[h][0][r] * inv);
      outp[(size_t)r * NC + 16] = (_Float16)(ov[h][1][r] * inv);
    }
  }
}

// --- Output projection + residual: out = x + W_proj @ A + b_proj -----------
__global__ __launch_bounds__(32)
void proj_kernel(const _Float16* __restrict__ Ah,  // [b][t][c] f16
                 const _Float16* __restrict__ wp,  // [128][128] f16
                 const float* __restrict__ bp,
                 const float* __restrict__ x,
                 float* __restrict__ out) {
  const int lane = threadIdx.x;
  const int n = lane & 15, g = lane >> 4;
  int blk = blockIdx.x;
  const int b  = blk / (8 * 256);
  blk %= (8 * 256);
  const int o0 = (blk >> 8) * 16;
  const int t0 = (blk & 255) * 16;

  v8f acc;
  #pragma unroll
  for (int r = 0; r < 8; ++r) acc[r] = bp[o0 + g * 8 + r];

  const _Float16* arow = Ah + ((size_t)b * TT + t0 + n) * NC;
  #pragma unroll
  for (int kk = 0; kk < 4; ++kk) {
    Frag16 a;
    const unsigned* ap = (const unsigned*)(wp + (size_t)(o0 + n) * NC + kk * 32);
    #pragma unroll
    for (int i = 0; i < 8; ++i) {
      int h2 = i >> 2, q = i & 3;
      a.u[i] = ap[g * 8 + h2 * 4 + q];
    }
    Frag16 bf;
    const uint4* bpv = (const uint4*)(arow + kk * 32 + g * 16);
    bf.q[0] = bpv[0];
    bf.q[1] = bpv[1];
    acc = __builtin_amdgcn_wmma_f32_16x16x32_f16(false, a.v, false, bf.v,
                                                 (short)0, acc, false, false);
  }

  #pragma unroll
  for (int r = 0; r < 8; ++r) {
    size_t idx = ((size_t)(b * NC + o0 + g * 8 + r)) * TT + t0 + n;
    out[idx] = x[idx] + acc[r];
  }
}

// ---------------------------------------------------------------------------
extern "C" void kernel_launch(void* const* d_in, const int* in_sizes, int n_in,
                              void* d_out, int out_size, void* d_ws, size_t ws_size,
                              hipStream_t stream) {
  const float* x      = (const float*)d_in[0];
  const float* w_qkv  = (const float*)d_in[1];
  const float* b_qkv  = (const float*)d_in[2];
  const float* w_proj = (const float*)d_in[3];
  const float* b_proj = (const float*)d_in[4];
  float* out = (float*)d_out;

  // workspace layout (bytes, 256-aligned); total ~10.6 MB
  char* ws = (char*)d_ws;
  _Float16* xT  = (_Float16*)(ws + 0);          // [b][t][c]  2,097,152
  _Float16* wqh = (_Float16*)(ws + 2097152);    //               98,304
  _Float16* wph = (_Float16*)(ws + 2195456);    //               32,768
  _Float16* Qb  = (_Float16*)(ws + 2228224);    // [bh][t][ch] 2,097,152
  _Float16* Kb  = (_Float16*)(ws + 4325376);    // [bh][t][ch] 2,097,152
  _Float16* Vb  = (_Float16*)(ws + 6422528);    // [bh][ch][t] 2,097,152
  _Float16* Ah  = (_Float16*)(ws + 8519680);    // [b][t][c]   2,097,152

  transpose_x_kernel<<<NB * (TT / 32) * (NC / 32), dim3(32, 8), 0, stream>>>(x, xT);
  cvt_f32_f16_kernel<<<(49152 + 255) / 256, 256, 0, stream>>>(w_qkv, wqh, 49152);
  cvt_f32_f16_kernel<<<(16384 + 255) / 256, 256, 0, stream>>>(w_proj, wph, 16384);

  qkv_kernel<<<NB * 24 * 256, 32, 0, stream>>>(xT, wqh, b_qkv, Qb, Kb, Vb);
  attn_kernel<<<(BH * 128) / 4, 128, 0, stream>>>(Qb, Kb, Vb, Ah);
  proj_kernel<<<NB * 8 * 256, 32, 0, stream>>>(Ah, wph, b_proj, x, out);
}
